// SAGEExtractor_57921928954039
// MI455X (gfx1250) — compile-verified, matrix operand order checked by reading, not run on hardware
//
#include <hip/hip_runtime.h>
#include <hip/hip_bf16.h>

// ---------------------------------------------------------------------------
// GraphSAGE layer for MI455X (gfx1250, wave32, WMMA).
//   agg = segment_mean(x[src] -> dst)
//   out = relu(agg @ W_l^T + b_l + x @ W_r^T) + x
// Pipeline:
//   1) zero agg+counts
//   2) degree counts        (global_atomic_add u32)
//   3) scatter-add msgs     (global_atomic_add f32; x L2-resident: 102MB < 192MB L2)
//   4) pack [W_l|W_r] -> Bcat bf16 [256 x 512]   (B^T row-major == W row-major)
//   5) convert [agg*inv | x] -> A_bf bf16 [N x 512]  (single streaming pass;
//      all fp32->bf16 conversion hoisted OUT of the GEMM)
//   6) pure-bf16 WMMA GEMM K=512, wave tile 16x32 (2 acc), epilogue
//      +bias, relu, +x residual.  Hot loop = loads + v_wmma only.
// ---------------------------------------------------------------------------

#define D 256
#define K2 512   // concatenated K

typedef __attribute__((ext_vector_type(16))) __bf16          v16bf;
typedef __attribute__((ext_vector_type(8)))  float           v8f;
typedef __attribute__((ext_vector_type(8)))  unsigned short  us8;

__device__ __forceinline__ unsigned short f2bf(float f) {
    // round-to-nearest-even fp32 -> bf16
    unsigned u = __float_as_uint(f);
    u += 0x7FFFu + ((u >> 16) & 1u);
    return (unsigned short)(u >> 16);
}

__device__ __forceinline__ ushort4 cvt4(float4 v, float sc) {
    return make_ushort4(f2bf(v.x * sc), f2bf(v.y * sc), f2bf(v.z * sc), f2bf(v.w * sc));
}

// --- 1) zero workspace (agg floats + counts ints, contiguous) ---------------
__global__ void sage_zero(float* __restrict__ p, long n) {
    long i = (long)blockIdx.x * blockDim.x + threadIdx.x;
    if (i < n) p[i] = 0.0f;
}

// --- 2) in-degree counts ----------------------------------------------------
__global__ void sage_count(const long long* __restrict__ ei, int* __restrict__ counts, int E) {
    int e = blockIdx.x * blockDim.x + threadIdx.x;
    if (e < E) atomicAdd(&counts[(int)ei[(long)E + e]], 1);
}

// --- 3) scatter-add messages: agg[dst] += x[src] ----------------------------
__global__ void sage_scatter(const long long* __restrict__ ei,
                             const float* __restrict__ x,
                             float* __restrict__ agg, int E) {
    long gid = (long)blockIdx.x * blockDim.x + threadIdx.x;
    if (gid >= (long)E * 64) return;
    int e = (int)(gid >> 6);
    int f = ((int)gid & 63) << 2;
    int src = (int)ei[e];
    int dst = (int)ei[(long)E + e];
    float4 v = *(const float4*)(x + (long)src * D + f);
    float* a = agg + (long)dst * D + f;
    atomicAdd(a + 0, v.x);
    atomicAdd(a + 1, v.y);
    atomicAdd(a + 2, v.z);
    atomicAdd(a + 3, v.w);
}

// --- 4) pack weights: Bcat[n][0:256]=W_l[n][:], Bcat[n][256:512]=W_r[n][:] --
__global__ void sage_pack(const float* __restrict__ Wl, const float* __restrict__ Wr,
                          unsigned short* __restrict__ B) {
    int gid = blockIdx.x * blockDim.x + threadIdx.x;   // 0 .. 256*512/4-1
    int n = gid >> 7;           // output row
    int f = (gid & 127) << 2;   // k' chunk
    float4 w = (f < D) ? *(const float4*)(Wl + n * D + f)
                       : *(const float4*)(Wr + n * D + (f - D));
    *(ushort4*)(B + (long)n * K2 + f) = cvt4(w, 1.0f);
}

// --- 5) convert A once: A_bf[i][0:256]=agg[i]*inv, A_bf[i][256:512]=x[i] ----
__global__ void sage_convert(const float* __restrict__ agg, const float* __restrict__ x,
                             const int* __restrict__ counts,
                             unsigned short* __restrict__ A, int Nrows) {
    long gid = (long)blockIdx.x * blockDim.x + threadIdx.x;
    if (gid >= (long)Nrows * 64) return;
    int i = (int)(gid >> 6);
    int f = ((int)gid & 63) << 2;
    int cnt = counts[i];
    float inv = 1.0f / (float)(cnt > 1 ? cnt : 1);
    float4 a  = *(const float4*)(agg + (long)i * D + f);
    float4 xv = *(const float4*)(x   + (long)i * D + f);
    *(ushort4*)(A + (long)i * K2 + f)     = cvt4(a, inv);
    *(ushort4*)(A + (long)i * K2 + D + f) = cvt4(xv, 1.0f);
}

// --- 6) pure bf16 WMMA GEMM + epilogue --------------------------------------
// Block = 256 threads = 8 waves; block tile 64(M) x 64(N).
// Wave tile 16(M) x 32(N): one A fragment feeds two B fragments per K-step.
__global__ __launch_bounds__(256) void sage_wmma_gemm(
    const unsigned short* __restrict__ A,   // [N x 512] bf16
    const unsigned short* __restrict__ B,   // [256 x 512] bf16 (B^T row-major)
    const float* __restrict__ bias,
    const float* __restrict__ x,
    float* __restrict__ out, int Nrows)
{
    const int lane = threadIdx.x & 31;
    const int wid  = threadIdx.x >> 5;
    const int tM   = blockIdx.x * 64 + (wid >> 1) * 16;
    const int tN0  = blockIdx.y * 64 + (wid & 1) * 32;   // two 16-col tiles
    const int half = lane >> 4;       // 0: K{0-7,16-23}  1: K{8-15,24-31}
    const int l15  = lane & 15;
    const int kb   = half * 8;

    int arow = tM + l15;
    int arow_c = arow < Nrows ? arow : (Nrows - 1);
    const unsigned short* aRow  = A + (long)arow_c * K2;
    const unsigned short* b0Row = B + (long)(tN0 + l15) * K2;
    const unsigned short* b1Row = B + (long)(tN0 + 16 + l15) * K2;

    union V16 { v16bf v; unsigned short s[16]; };
    v8f acc0 = {}, acc1 = {};

#pragma unroll 4
    for (int k0 = 0; k0 < K2; k0 += 32) {
        const int ka = k0 + kb;
        V16 Af, B0, B1;
        *(us8*)&Af.s[0] = *(const us8*)(aRow + ka);
        *(us8*)&Af.s[8] = *(const us8*)(aRow + ka + 16);
        *(us8*)&B0.s[0] = *(const us8*)(b0Row + ka);
        *(us8*)&B0.s[8] = *(const us8*)(b0Row + ka + 16);
        *(us8*)&B1.s[0] = *(const us8*)(b1Row + ka);
        *(us8*)&B1.s[8] = *(const us8*)(b1Row + ka + 16);

        acc0 = __builtin_amdgcn_wmma_f32_16x16x32_bf16(
                   false, Af.v, false, B0.v, (short)0, acc0, false, false);
        acc1 = __builtin_amdgcn_wmma_f32_16x16x32_bf16(
                   false, Af.v, false, B1.v, (short)0, acc1, false, false);
    }

    // Epilogue. C/D layout: VGPR r -> M = r + 8*half, N = lane & 15.
    const int col0 = tN0 + l15;
    const int col1 = tN0 + 16 + l15;
    const float bb0 = bias[col0];
    const float bb1 = bias[col1];
#pragma unroll
    for (int r = 0; r < 8; ++r) {
        int row = tM + r + half * 8;
        if (row < Nrows) {
            long base = (long)row * D;
            float v0 = fmaxf(acc0[r] + bb0, 0.0f);
            float v1 = fmaxf(acc1[r] + bb1, 0.0f);
            out[base + col0] = v0 + x[base + col0];   // residual
            out[base + col1] = v1 + x[base + col1];
        }
    }
}

// ---------------------------------------------------------------------------
extern "C" void kernel_launch(void* const* d_in, const int* in_sizes, int n_in,
                              void* d_out, int out_size, void* d_ws, size_t ws_size,
                              hipStream_t stream) {
    const float*     x  = (const float*)d_in[0];
    const long long* ei = (const long long*)d_in[1];   // int64 [2,E]
    // d_in[2] = edge_weight (unused by reference message())
    const float*     Wl = (const float*)d_in[3];
    const float*     bl = (const float*)d_in[4];
    const float*     Wr = (const float*)d_in[5];
    float* out = (float*)d_out;

    const int N = in_sizes[0] / D;
    const int E = in_sizes[2];

    // workspace layout
    char* ws = (char*)d_ws;
    size_t offAgg = 0;
    size_t offCnt = offAgg + (size_t)N * D * sizeof(float);
    size_t offA   = (offCnt + (size_t)N * sizeof(int) + 255) & ~(size_t)255;
    size_t offB   = offA + (size_t)N * K2 * sizeof(unsigned short);
    float*          agg    = (float*)(ws + offAgg);
    int*            counts = (int*)(ws + offCnt);
    unsigned short* Abf    = (unsigned short*)(ws + offA);
    unsigned short* Bcat   = (unsigned short*)(ws + offB);

    // 1) zero agg + counts (contiguous; 0.0f bit pattern == int 0)
    long nz = (long)N * D + N;
    sage_zero<<<(unsigned)((nz + 255) / 256), 256, 0, stream>>>(agg, nz);

    // 2) degrees
    sage_count<<<(E + 255) / 256, 256, 0, stream>>>(ei, counts, E);

    // 3) scatter-add messages
    long nsc = (long)E * 64;
    sage_scatter<<<(unsigned)((nsc + 255) / 256), 256, 0, stream>>>(ei, x, agg, E);

    // 4) pack weights into Bcat [256 x 512] bf16
    sage_pack<<<(D * K2 / 4) / 256, 256, 0, stream>>>(Wl, Wr, Bcat);

    // 5) convert A once into A_bf [N x 512] bf16
    long ncv = (long)N * 64;
    sage_convert<<<(unsigned)((ncv + 255) / 256), 256, 0, stream>>>(x ? agg : agg, x, counts, Abf, N);

    // 6) pure-bf16 WMMA GEMM + bias + relu + residual
    dim3 grid((N + 63) / 64, D / 64);
    sage_wmma_gemm<<<grid, 256, 0, stream>>>(Abf, Bcat, bl, x, out, N);
}
